// VisionCrossAttentionLayer_13185549599085
// MI455X (gfx1250) — compile-verified, matrix operand order checked by reading, not run on hardware
//
#include <hip/hip_runtime.h>
#include <hip/hip_bf16.h>
#include <math.h>

// ---------------------------------------------------------------------------
// VisionCrossAttentionLayer for MI455X (gfx1250, wave32, WMMA bf16)
//   B=16, D=1024, G=576, L=64, QL=16, V=1024, H=16, hd=64, Wt=36
// All heavy math runs through v_wmma_f32_16x16x32_bf16 (fp32 accumulate).
// ---------------------------------------------------------------------------

typedef __attribute__((ext_vector_type(16))) __bf16          v16bf;
typedef __attribute__((ext_vector_type(8)))  float           v8f;
typedef __attribute__((ext_vector_type(8)))  unsigned short  ushort8v;
typedef __attribute__((ext_vector_type(16))) unsigned short  ushort16v;
typedef __attribute__((ext_vector_type(4)))  unsigned short  ushort4v;

__device__ __forceinline__ unsigned short f2bf(float f) {
  unsigned int u = __float_as_uint(f);
  return (unsigned short)((u + 0x7FFFu + ((u >> 16) & 1u)) >> 16);  // RNE
}

__device__ __forceinline__ float gelu_exact(float x) {
  return 0.5f * x * (1.0f + erff(x * 0.70710678118654752f));
}

// WMMA A/B fragment gather from an LDS tile staged as bf16 bits, K-contiguous
// rows. Caller passes pointer to (row*stride + kchunk) where
//   row    = tile_row_base + (lane & 15)
//   kchunk = kstep_base + (lane >> 4) * 8
// Per ISA 7.12.2: elements 0..7 = K[kchunk..+7], elements 8..15 = K[kchunk+16..+23].
__device__ __forceinline__ v16bf frag_ld(const unsigned short* p) {
  ushort8v lo = *(const ushort8v*)(p);
  ushort8v hi = *(const ushort8v*)(p + 16);
  ushort16v u;
#pragma unroll
  for (int i = 0; i < 8; ++i) { u[i] = lo[i]; u[i + 8] = hi[i]; }
  return __builtin_bit_cast(v16bf, u);
}

// window gather remap: row r in (B, QL=16, Wt=36) order -> row in (B, G=576)
__device__ __forceinline__ int win_remap(int r) {
  int b = r / 576; int t = r - b * 576;
  int q = t / 36;  int w = t - q * 36;
  int a = q >> 2, c = q & 3;
  int bb = w / 6, dd = w - bb * 6;
  return b * 576 + a * 144 + bb * 24 + c * 6 + dd;
}

// ---------------------------------------------------------------------------
// Generic GEMM:  C[M,N] = Epilogue( LN?(A)[M,K] * W[N,K]^T )
//   A1 != null : A columns are concat(A0, A1) split at Ksplit (row len Ksplit each)
//   lnStats    : per-row {mean, rstd}; lnG/lnBeta gamma/beta over K (fused LN)
//   remapWin   : A row index remapped through win_remap (window extraction)
//   EPI: 0 = none, 1 = C = acc + addSrc, 2 = C = gelu(acc)
// Block tile 128x128, BK=32, 256 threads = 8 waves (4 M x 2 N),
// each wave owns a 32x64 sub-tile = 2x4 WMMA tiles.
// ---------------------------------------------------------------------------
template <int EPI>
__global__ __launch_bounds__(256, 2)
void gemm_wmma_bf16(const float* __restrict__ A0, const float* __restrict__ A1,
                    int Ksplit, const float* __restrict__ Bw, float* C,
                    int M, int N, int K,
                    const float* __restrict__ lnStats, const float* __restrict__ lnG,
                    const float* __restrict__ lnBeta, const float* addSrc,
                    int remapWin) {
  constexpr int BM = 128, BN = 128, BK = 32, LDT = 40;  // LDT: padded LDS row stride
  __shared__ __align__(16) unsigned short As[BM * LDT];
  __shared__ __align__(16) unsigned short Bs[BN * LDT];

  const int tid = threadIdx.x;
  const int lane = tid & 31, wave = tid >> 5;
  const int waveM = wave & 3, waveN = wave >> 2;
  const int m0 = blockIdx.y * BM, n0 = blockIdx.x * BN;

  const v8f vzero = {0.f, 0.f, 0.f, 0.f, 0.f, 0.f, 0.f, 0.f};
  v8f acc[2][4];
#pragma unroll
  for (int mt = 0; mt < 2; ++mt)
#pragma unroll
    for (int nt = 0; nt < 4; ++nt) acc[mt][nt] = vzero;

  for (int k0 = 0; k0 < K; k0 += BK) {
    const float* Asrc = A0; int kk = k0; int lda = K;
    if (A1) { lda = Ksplit; if (k0 >= Ksplit) { Asrc = A1; kk = k0 - Ksplit; } }

    // ---- stage A tile (128 x 32 fp32 -> bf16 in LDS) ----
#pragma unroll
    for (int i = 0; i < 4; ++i) {
      int s = tid + i * 256;
      int row = s >> 3, c4 = s & 7;
      int srow = m0 + row;
      if (remapWin) srow = win_remap(srow);
      const float4 av = *(const float4*)(Asrc + (size_t)srow * lda + kk + c4 * 4);
      float x0 = av.x, x1 = av.y, x2 = av.z, x3 = av.w;
      if (lnG) {
        float mu = lnStats[(size_t)(m0 + row) * 2];
        float rs = lnStats[(size_t)(m0 + row) * 2 + 1];
        const float4 g  = *(const float4*)(lnG + k0 + c4 * 4);
        const float4 be = *(const float4*)(lnBeta + k0 + c4 * 4);
        x0 = (x0 - mu) * rs * g.x + be.x;
        x1 = (x1 - mu) * rs * g.y + be.y;
        x2 = (x2 - mu) * rs * g.z + be.z;
        x3 = (x3 - mu) * rs * g.w + be.w;
      }
      ushort4v w4;
      w4[0] = f2bf(x0); w4[1] = f2bf(x1); w4[2] = f2bf(x2); w4[3] = f2bf(x3);
      *(ushort4v*)(As + row * LDT + c4 * 4) = w4;
    }
    // ---- stage B tile (weights, 128 x 32) ----
#pragma unroll
    for (int i = 0; i < 4; ++i) {
      int s = tid + i * 256;
      int row = s >> 3, c4 = s & 7;
      const float4 bv = *(const float4*)(Bw + (size_t)(n0 + row) * K + k0 + c4 * 4);
      ushort4v w4;
      w4[0] = f2bf(bv.x); w4[1] = f2bf(bv.y); w4[2] = f2bf(bv.z); w4[3] = f2bf(bv.w);
      *(ushort4v*)(Bs + row * LDT + c4 * 4) = w4;
    }
    // prefetch next k-step (gfx1250 global_prefetch path)
    if (k0 + BK < K) {
      int prow = tid >> 3;
      __builtin_prefetch(Bw + (size_t)(n0 + prow) * K + k0 + BK, 0, 1);
      if (!A1) {
        int psrow = m0 + prow;
        if (remapWin) psrow = win_remap(psrow);
        __builtin_prefetch(Asrc + (size_t)psrow * lda + kk + BK, 0, 1);
      }
    }
    __syncthreads();

    // ---- WMMA compute ----
    v16bf afr[2], bfr[4];
#pragma unroll
    for (int mt = 0; mt < 2; ++mt)
      afr[mt] = frag_ld(As + (waveM * 32 + mt * 16 + (lane & 15)) * LDT + ((lane >> 4) * 8));
#pragma unroll
    for (int nt = 0; nt < 4; ++nt)
      bfr[nt] = frag_ld(Bs + (waveN * 64 + nt * 16 + (lane & 15)) * LDT + ((lane >> 4) * 8));
#pragma unroll
    for (int mt = 0; mt < 2; ++mt)
#pragma unroll
      for (int nt = 0; nt < 4; ++nt)
        acc[mt][nt] = __builtin_amdgcn_wmma_f32_16x16x32_bf16(
            false, afr[mt], false, bfr[nt], (short)0, acc[mt][nt], false, false);
    __syncthreads();
  }

  // ---- epilogue: C-layout lane 0-15 N=lane, lane 16-31 N=lane-16; vgpr j: M=j(+8) ----
#pragma unroll
  for (int mt = 0; mt < 2; ++mt)
#pragma unroll
    for (int nt = 0; nt < 4; ++nt)
#pragma unroll
      for (int j = 0; j < 8; ++j) {
        int m = m0 + waveM * 32 + mt * 16 + j + ((lane >> 4) << 3);
        int n = n0 + waveN * 64 + nt * 16 + (lane & 15);
        float v = acc[mt][nt][j];
        if (EPI == 2) v = gelu_exact(v);
        if (EPI == 1) v += addSrc[(size_t)m * N + n];
        C[(size_t)m * N + n] = v;
      }
}

// ---------------------------------------------------------------------------
// Per-row LayerNorm statistics (mean, rstd) for rows of length 1024.
// ---------------------------------------------------------------------------
__global__ __launch_bounds__(256)
void row_stats_kernel(const float* __restrict__ X, float* __restrict__ stats) {
  __shared__ float sh[512];
  const int row = blockIdx.x, tid = threadIdx.x;
  const float* xr = X + (size_t)row * 1024;
  float s = 0.f, s2 = 0.f;
#pragma unroll
  for (int i = 0; i < 4; ++i) { float v = xr[tid + i * 256]; s += v; s2 += v * v; }
  sh[tid] = s; sh[256 + tid] = s2;
  __syncthreads();
  for (int off = 128; off > 0; off >>= 1) {
    if (tid < off) { sh[tid] += sh[tid + off]; sh[256 + tid] += sh[256 + tid + off]; }
    __syncthreads();
  }
  if (tid == 0) {
    float mu = sh[0] * (1.f / 1024.f);
    float var = sh[256] * (1.f / 1024.f) - mu * mu;
    stats[(size_t)row * 2] = mu;
    stats[(size_t)row * 2 + 1] = rsqrtf(var + 1e-5f);
  }
}

// ---------------------------------------------------------------------------
// Windowed cross attention. One block per (b, qi, h): 4096 blocks, 128 thr.
//   Q tile 64(L) x 64(hd); K/V: 36 keys padded to 64.
//   S = Q K^T (WMMA) -> masked softmax (LDS) -> P V (WMMA, V transposed in LDS)
// ---------------------------------------------------------------------------
__global__ __launch_bounds__(128)
void attn_win_kernel(const float* __restrict__ Qb, const float* __restrict__ Kb,
                     const float* __restrict__ Vb, float* __restrict__ Ob) {
  constexpr int LQ = 72;  // padded bf16 row stride (16B-aligned rows)
  __shared__ __align__(16) unsigned short Qs[64 * LQ];
  __shared__ __align__(16) unsigned short Ks[64 * LQ];
  __shared__ __align__(16) unsigned short Vt[64 * LQ];
  __shared__ __align__(16) unsigned short At[64 * LQ];
  __shared__ float Ss[64 * 65];

  const int tid = threadIdx.x, lane = tid & 31, wv = tid >> 5;
  const int bid = blockIdx.x;
  const int h = bid & 15, qi = (bid >> 4) & 15, b = bid >> 8;
  const size_t qbase = ((size_t)b * 1024 + qi) * 1024 + (size_t)h * 64;  // + l*16*1024
  const size_t kbase = ((size_t)b * 576 + qi * 36) * 1024 + (size_t)h * 64;

  // ---- stage Q (64x64), K (pad to 64 rows), V transposed ----
#pragma unroll
  for (int i = 0; i < 8; ++i) {
    int s = tid + i * 128;
    int r = s >> 4, c4 = s & 15;
    float4 qv = *(const float4*)(Qb + qbase + (size_t)r * 16 * 1024 + c4 * 4);
    ushort4v q4;
    q4[0] = f2bf(qv.x); q4[1] = f2bf(qv.y); q4[2] = f2bf(qv.z); q4[3] = f2bf(qv.w);
    *(ushort4v*)(Qs + r * LQ + c4 * 4) = q4;

    float4 kv = {0.f, 0.f, 0.f, 0.f};
    if (r < 36) kv = *(const float4*)(Kb + kbase + (size_t)r * 1024 + c4 * 4);
    ushort4v k4;
    k4[0] = f2bf(kv.x); k4[1] = f2bf(kv.y); k4[2] = f2bf(kv.z); k4[3] = f2bf(kv.w);
    *(ushort4v*)(Ks + r * LQ + c4 * 4) = k4;

    float4 vv = {0.f, 0.f, 0.f, 0.f};
    if (r < 36) vv = *(const float4*)(Vb + kbase + (size_t)r * 1024 + c4 * 4);
    Vt[(c4 * 4 + 0) * LQ + r] = f2bf(vv.x);
    Vt[(c4 * 4 + 1) * LQ + r] = f2bf(vv.y);
    Vt[(c4 * 4 + 2) * LQ + r] = f2bf(vv.z);
    Vt[(c4 * 4 + 3) * LQ + r] = f2bf(vv.w);
  }
  __syncthreads();

  const v8f vzero = {0.f, 0.f, 0.f, 0.f, 0.f, 0.f, 0.f, 0.f};

  // ---- S = Q K^T : wave wv owns rows [16wv, 16wv+16) ----
  v8f sacc[4] = {vzero, vzero, vzero, vzero};
#pragma unroll
  for (int ks = 0; ks < 2; ++ks) {
    v16bf aq = frag_ld(Qs + (wv * 16 + (lane & 15)) * LQ + ks * 32 + (lane >> 4) * 8);
#pragma unroll
    for (int nt = 0; nt < 4; ++nt) {
      v16bf bk = frag_ld(Ks + (nt * 16 + (lane & 15)) * LQ + ks * 32 + (lane >> 4) * 8);
      sacc[nt] = __builtin_amdgcn_wmma_f32_16x16x32_bf16(
          false, aq, false, bk, (short)0, sacc[nt], false, false);
    }
  }
#pragma unroll
  for (int nt = 0; nt < 4; ++nt)
#pragma unroll
    for (int j = 0; j < 8; ++j) {
      int m = wv * 16 + j + ((lane >> 4) << 3);
      int n = nt * 16 + (lane & 15);
      Ss[m * 65 + n] = sacc[nt][j];
    }
  __syncthreads();

  // ---- masked softmax over 36 valid keys (scale = 1/sqrt(64)) ----
  if (tid < 64) {
    const float scale = 0.125f;
    float mx = -3.4e38f;
    for (int w = 0; w < 36; ++w) mx = fmaxf(mx, Ss[tid * 65 + w] * scale);
    float den = 0.f;
    for (int w = 0; w < 36; ++w) {
      float e = __expf(Ss[tid * 65 + w] * scale - mx);
      Ss[tid * 65 + w] = e;
      den += e;
    }
    float inv = 1.f / den;
    for (int w = 0; w < 64; ++w)
      At[tid * LQ + w] = f2bf((w < 36) ? Ss[tid * 65 + w] * inv : 0.f);
  }
  __syncthreads();

  // ---- O = P V ----
  v8f oacc[4] = {vzero, vzero, vzero, vzero};
#pragma unroll
  for (int ks = 0; ks < 2; ++ks) {
    v16bf ap = frag_ld(At + (wv * 16 + (lane & 15)) * LQ + ks * 32 + (lane >> 4) * 8);
#pragma unroll
    for (int nt = 0; nt < 4; ++nt) {
      v16bf bv = frag_ld(Vt + (nt * 16 + (lane & 15)) * LQ + ks * 32 + (lane >> 4) * 8);
      oacc[nt] = __builtin_amdgcn_wmma_f32_16x16x32_bf16(
          false, ap, false, bv, (short)0, oacc[nt], false, false);
    }
  }
#pragma unroll
  for (int nt = 0; nt < 4; ++nt)
#pragma unroll
    for (int j = 0; j < 8; ++j) {
      int m = wv * 16 + j + ((lane >> 4) << 3);  // l in [0,64)
      int n = nt * 16 + (lane & 15);             // e in [0,64)
      Ob[((size_t)b * 1024 + m * 16 + qi) * 1024 + (size_t)h * 64 + n] = oacc[nt][j];
    }
}

// ---------------------------------------------------------------------------
extern "C" void kernel_launch(void* const* d_in, const int* in_sizes, int n_in,
                              void* d_out, int out_size, void* d_ws, size_t ws_size,
                              hipStream_t stream) {
  (void)in_sizes; (void)n_in; (void)out_size; (void)ws_size;
  const float* global_ctx   = (const float*)d_in[1];
  const float* queries      = (const float*)d_in[4];
  const float* image_tokens = (const float*)d_in[5];
  const float* W_in  = (const float*)d_in[7];
  const float* W_ctx = (const float*)d_in[8];
  const float* ln_q_g = (const float*)d_in[9],  *ln_q_b = (const float*)d_in[10];
  const float* Wq = (const float*)d_in[11];
  const float* ln_k_g = (const float*)d_in[12], *ln_k_b = (const float*)d_in[13];
  const float* Wk = (const float*)d_in[14];
  const float* ln_v_g = (const float*)d_in[15], *ln_v_b = (const float*)d_in[16];
  const float* Wv = (const float*)d_in[17];
  const float* Wo = (const float*)d_in[18];
  const float* norm_g = (const float*)d_in[19], *norm_b = (const float*)d_in[20];
  const float* W1 = (const float*)d_in[21];
  const float* W2 = (const float*)d_in[22];
  float* out = (float*)d_out;

  const size_t SZ16 = (size_t)16384 * 1024;  // B*V rows
  const size_t SZ9  = (size_t)9216 * 1024;   // B*G rows
  float* ws   = (float*)d_ws;
  float* pq   = ws;                 // (B*V, D); later xpre (in-place residual)
  float* qb   = pq + SZ16;          // q; later h1
  float* ao   = qb + SZ16;          // attention output
  float* winP = ao + SZ16;          // projected windows (B*G, D)
  float* kb   = winP + SZ9;
  float* vb   = kb + SZ9;
  float* stA  = vb + SZ9;           // 16384 x {mean,rstd}
  float* stW  = stA + 16384 * 2;    // 9216  x {mean,rstd}

  const dim3 blk(256);
  const dim3 gBig(8, 128);  // N=1024, M=16384
  const dim3 gWin(8, 72);   // N=1024, M=9216

  // 1) pq = concat(queries, image_tokens) @ W_in^T
  gemm_wmma_bf16<0><<<gBig, blk, 0, stream>>>(queries, image_tokens, 1024, W_in, pq,
      16384, 1024, 2048, nullptr, nullptr, nullptr, nullptr, 0);
  // 2) winP = window_gather(global_ctx) @ W_ctx^T   (row remap fused into loader)
  gemm_wmma_bf16<0><<<gWin, blk, 0, stream>>>(global_ctx, nullptr, 0, W_ctx, winP,
      9216, 1024, 1024, nullptr, nullptr, nullptr, nullptr, 1);
  // 3) LN stats
  row_stats_kernel<<<16384, blk, 0, stream>>>(pq, stA);
  row_stats_kernel<<<9216, blk, 0, stream>>>(winP, stW);
  // 4) q/k/v projections with fused LayerNorm on A
  gemm_wmma_bf16<0><<<gBig, blk, 0, stream>>>(pq, nullptr, 0, Wq, qb,
      16384, 1024, 1024, stA, ln_q_g, ln_q_b, nullptr, 0);
  gemm_wmma_bf16<0><<<gWin, blk, 0, stream>>>(winP, nullptr, 0, Wk, kb,
      9216, 1024, 1024, stW, ln_k_g, ln_k_b, nullptr, 0);
  gemm_wmma_bf16<0><<<gWin, blk, 0, stream>>>(winP, nullptr, 0, Wv, vb,
      9216, 1024, 1024, stW, ln_v_g, ln_v_b, nullptr, 0);
  // 5) windowed attention
  attn_win_kernel<<<4096, dim3(128), 0, stream>>>(qb, kb, vb, ao);
  // 6) xpre = ao @ Wo^T + pq  (written in-place over pq; 1:1 element read->write)
  gemm_wmma_bf16<1><<<gBig, blk, 0, stream>>>(ao, nullptr, 0, Wo, pq,
      16384, 1024, 1024, nullptr, nullptr, nullptr, pq, 0);
  // 7) stats for final LN, then h1 = gelu(LN(xpre) @ W1^T)
  row_stats_kernel<<<16384, blk, 0, stream>>>(pq, stA);
  gemm_wmma_bf16<2><<<gBig, blk, 0, stream>>>(pq, nullptr, 0, W1, qb,
      16384, 1024, 1024, stA, norm_g, norm_b, nullptr, 0);
  // 8) out = h1 @ W2^T + image_tokens
  gemm_wmma_bf16<1><<<gBig, blk, 0, stream>>>(qb, nullptr, 0, W2, out,
      16384, 1024, 1024, nullptr, nullptr, nullptr, image_tokens, 0);
}
